// AmplifyModel_3496103379438
// MI455X (gfx1250) — matrix operand order repaired and checked by reading, not compiled, not run on hardware
//
#include <hip/hip_runtime.h>
#include <hip/hip_bf16.h>

typedef __attribute__((ext_vector_type(16))) _Float16 v16h;
typedef __attribute__((ext_vector_type(8)))  _Float16 v8h;
typedef __attribute__((ext_vector_type(8)))  float    v8f;

#define NPTS    100000
#define NEDGE   800000
#define CH      128
#define NLAYERS 8
#define BN_EPS  1e-5f

__device__ __forceinline__ float elu_f(float x) {
    return x > 0.f ? x : (__expf(x) - 1.f);
}

// ---------------------------------------------------------------------------
// zero helper
// ---------------------------------------------------------------------------
__global__ void zero_f32(float* __restrict__ p, long long n) {
    long long i = (long long)blockIdx.x * blockDim.x + threadIdx.x;
    if (i < n) p[i] = 0.f;
}

// ---------------------------------------------------------------------------
// embed: X[n,c] = sum_j inputs[n,j] * W_in[j,c] + b_in[c]
// ---------------------------------------------------------------------------
__global__ void embed_kernel(const float* __restrict__ inp,
                             const float* __restrict__ Win,
                             const float* __restrict__ bin,
                             float* __restrict__ X, int N) {
    long long idx = (long long)blockIdx.x * blockDim.x + threadIdx.x;
    if (idx >= (long long)N * CH) return;
    int n = (int)(idx >> 7);
    int c = (int)(idx & (CH - 1));
    float a0 = inp[(size_t)n * 3 + 0];
    float a1 = inp[(size_t)n * 3 + 1];
    float a2 = inp[(size_t)n * 3 + 2];
    X[idx] = a0 * Win[0 * CH + c] + a1 * Win[1 * CH + c] + a2 * Win[2 * CH + c] + bin[c];
}

// ---------------------------------------------------------------------------
// SpMM with fused ELU: oph[rows[e], c] += vals[e] * elu(X[cols[e], c])
// ---------------------------------------------------------------------------
__global__ void spmm_edges(const int* __restrict__ rows, const int* __restrict__ cols,
                           const float* __restrict__ vals, const float* __restrict__ X,
                           float* __restrict__ oph, long long total) {
    long long idx = (long long)blockIdx.x * blockDim.x + threadIdx.x;
    if (idx >= total) return;
    int e = (int)(idx >> 7);
    int c = (int)(idx & (CH - 1));
    float v = vals[e] * elu_f(X[(size_t)cols[e] * CH + c]);
    atomicAdd(&oph[(size_t)rows[e] * CH + c], v);
}

// ---------------------------------------------------------------------------
// masked column-sum of elu(X) for global average (128 threads = channels)
// avgbuf[0..127] = sum_n elu(X[n,c])*mask[n], avgbuf[128] = sum_n mask[n]
// ---------------------------------------------------------------------------
__global__ void avg_reduce(const float* __restrict__ X, const float* __restrict__ mask,
                           float* __restrict__ avgbuf, int N, int rowsPerBlock) {
    int c = threadIdx.x;            // 0..127
    int r0 = blockIdx.x * rowsPerBlock;
    int r1 = r0 + rowsPerBlock; if (r1 > N) r1 = N;
    float s = 0.f, ms = 0.f;
    for (int r = r0; r < r1; ++r) {
        float m = mask[r];
        s += elu_f(X[(size_t)r * CH + c]) * m;
        if (c == 0) ms += m;
    }
    atomicAdd(&avgbuf[c], s);
    if (c == 0) atomicAdd(&avgbuf[CH], ms);
}

__global__ void avg_bcast(const float* __restrict__ avgbuf, float* __restrict__ oph, long long n) {
    long long i = (long long)blockIdx.x * blockDim.x + threadIdx.x;
    if (i >= n) return;
    int c = (int)(i & (CH - 1));
    oph[i] = avgbuf[c] / avgbuf[CH];
}

// ---------------------------------------------------------------------------
// Fused ELU + pack to f16 A[N,K] + per-channel sum/sumsq stats.
// K = 256: channels 0..127 = elu(src0), 128..255 = src1 (op output, no elu)
// K = 128: channels 0..127 = elu(src0) only.
// ---------------------------------------------------------------------------
__global__ void pack_stats(const float* __restrict__ src0, const float* __restrict__ src1,
                           _Float16* __restrict__ A16, float* __restrict__ stats,
                           int N, int K, int rowsPerBlock) {
    int c = threadIdx.x;
    if (c >= K) return;
    int r0 = blockIdx.x * rowsPerBlock;
    int r1 = r0 + rowsPerBlock; if (r1 > N) r1 = N;
    const bool lo = (c < CH);
    const float* src = lo ? src0 : src1;
    int cc = lo ? c : (c - CH);
    float s = 0.f, ss = 0.f;
    for (int r = r0; r < r1; ++r) {
        float v = src[(size_t)r * CH + cc];
        if (lo) v = elu_f(v);
        A16[(size_t)r * K + c] = (_Float16)v;
        s += v; ss += v * v;
    }
    atomicAdd(&stats[c], s);
    atomicAdd(&stats[K + c], ss);
}

// ---------------------------------------------------------------------------
// Fold batch-norm into weights:  y = BN(a)@W + b  ==  a @ (diag(s)W) + (b + t@W)
// with s_k = g_k * rsqrt(var_k + eps), t_k = bt_k - mean_k * s_k.
// Writes Wt16[col][k] (transposed, f16) padded to 128 cols, and biasOut[128].
// ---------------------------------------------------------------------------
__global__ void fold_weights(const float* __restrict__ W, const float* __restrict__ b,
                             const float* __restrict__ g, const float* __restrict__ bt,
                             const float* __restrict__ stats,
                             _Float16* __restrict__ Wt16, float* __restrict__ biasOut,
                             int K, int ncols, float invN) {
    int col = threadIdx.x;          // 0..127 (one block of 128 threads)
    if (col >= CH) return;
    float acc = (col < ncols) ? b[col] : 0.f;
    for (int k = 0; k < K; ++k) {
        float m   = stats[k] * invN;
        float var = stats[K + k] * invN - m * m;
        float s   = g[k] * rsqrtf(var + BN_EPS);
        float t   = bt[k] - m * s;
        float w   = (col < ncols) ? W[(size_t)k * ncols + col] : 0.f;
        Wt16[(size_t)col * K + k] = (_Float16)(w * s);
        acc += t * w;
    }
    biasOut[col] = acc;
}

// ---------------------------------------------------------------------------
// WMMA GEMM: Y[N, ldOut] = A16[N,K] @ Wt16[128 cols][K]^T + bias (+resid)(+inp3 tile)
// 256 threads = 8 waves per workgroup; each wave owns a 16-row x 128-col strip.
// A stream + output use non-temporal hints (read/written exactly once);
// weights stay regular-temporal so they remain resident in WGP$/L2.
// ---------------------------------------------------------------------------
__global__ __launch_bounds__(256)
void gemm_wmma(const _Float16* __restrict__ A16, const _Float16* __restrict__ Wt16,
               const float* __restrict__ biasv, const float* __restrict__ resid,
               const float* __restrict__ inp3, float* __restrict__ Y,
               int N, int K, int ncols, int ldOut) {
    const int wave = threadIdx.x >> 5;      // 0..7
    const int lane = threadIdx.x & 31;
    const int l16  = lane & 15;
    const int hi   = lane >> 4;             // 0/1 : half-wave select
    const int rowBase = blockIdx.x * 128 + wave * 16;

    // A-operand row for this lane (clamped for the ragged tail tile)
    int arow = rowBase + l16;
    if (arow >= N) arow = N - 1;
    const _Float16* __restrict__ Arow = A16 + (size_t)arow * K;

    v8f acc[8];
#pragma unroll
    for (int t = 0; t < 8; ++t) acc[t] = (v8f)0.f;

    for (int k = 0; k < K; k += 32) {
        // A 16x32 f16, contiguous K order matching B:
        // lanes 0-15 hold K {0..15}; lanes 16-31 hold K {16..31}
        v8h a0 = __builtin_nontemporal_load((const v8h*)(Arow + k + hi * 16));
        v8h a1 = __builtin_nontemporal_load((const v8h*)(Arow + k + hi * 16 + 8));
        v16h a = __builtin_shufflevector(a0, a1, 0,1,2,3,4,5,6,7,8,9,10,11,12,13,14,15);
#pragma unroll
        for (int t = 0; t < 8; ++t) {
            // B 32x16 f16: lane = output column, hi selects K 0-15 vs 16-31;
            // contiguous in the [col][k] transposed weight (cache-resident).
            const _Float16* __restrict__ Bcol = Wt16 + (size_t)(t * 16 + l16) * K + k + hi * 16;
            v16h bv = *(const v16h*)Bcol;
            acc[t] = __builtin_amdgcn_wmma_f32_16x16x32_f16(
                false, a, false, bv, (short)0, acc[t], false, false);
        }
    }

    // C/D layout: VGPR r -> row = rowBase + hi*8 + r, col = tile*16 + l16
#pragma unroll
    for (int t = 0; t < 8; ++t) {
        int col = t * 16 + l16;
        if (col >= ncols) continue;
        float bcol = biasv[col];
#pragma unroll
        for (int r = 0; r < 8; ++r) {
            int row = rowBase + hi * 8 + r;
            if (row < N) {
                float v = acc[t][r] + bcol;
                if (resid) v += resid[(size_t)row * CH + col];
                if (inp3)  v += inp3[(size_t)row * 3 + (col % 3)];
                __builtin_nontemporal_store(v, &Y[(size_t)row * ldOut + col]);
            }
        }
    }
}

// ---------------------------------------------------------------------------
// host side
// ---------------------------------------------------------------------------
static inline size_t align_up(size_t v, size_t a) { return (v + a - 1) & ~(a - 1); }

extern "C" void kernel_launch(void* const* d_in, const int* in_sizes, int n_in,
                              void* d_out, int out_size, void* d_ws, size_t ws_size,
                              hipStream_t stream) {
    (void)in_sizes; (void)n_in; (void)out_size; (void)ws_size;
    const int N = NPTS;

    const float* inputs = (const float*)d_in[0];
    const float* mask   = (const float*)d_in[1];
    const int*   Lrows  = (const int*)  d_in[2];
    const int*   Lcols  = (const int*)  d_in[3];
    const float* Lvals  = (const float*)d_in[4];
    const float* W_in   = (const float*)d_in[5];
    const float* b_in   = (const float*)d_in[6];
    const float* W0     = (const float*)d_in[7];
    const float* b0     = (const float*)d_in[8];
    const float* g0     = (const float*)d_in[9];
    const float* bt0    = (const float*)d_in[10];
    const float* W1     = (const float*)d_in[11];
    const float* b1     = (const float*)d_in[12];
    const float* g1     = (const float*)d_in[13];
    const float* bt1    = (const float*)d_in[14];
    const float* W_out  = (const float*)d_in[15];
    const float* b_out  = (const float*)d_in[16];
    const float* g_out  = (const float*)d_in[17];
    const float* bt_out = (const float*)d_in[18];
    float* out = (float*)d_out;

    // workspace carve
    size_t off = 0;
    char* base = (char*)d_ws;
    auto carve = [&](size_t bytes) -> char* {
        off = align_up(off, 256);
        char* p = base + off;
        off += bytes;
        return p;
    };
    float*    X       = (float*)   carve((size_t)N * CH * 4);
    float*    Yb      = (float*)   carve((size_t)N * CH * 4);
    float*    oph     = (float*)   carve((size_t)N * CH * 4);
    _Float16* A16     = (_Float16*)carve((size_t)N * 2 * CH * 2);
    _Float16* Wt16    = (_Float16*)carve((size_t)CH * 2 * CH * 2);
    float*    biasBuf = (float*)   carve(CH * 4);
    float*    stats   = (float*)   carve(2 * 2 * CH * 4);
    float*    avgbuf  = (float*)   carve((CH + 1) * 4);

    const long long NC = (long long)N * CH;
    const int bThr = 256;
    const int gNC  = (int)((NC + bThr - 1) / bThr);
    const int rowsPerBlk = 256;
    const int gRows = (N + rowsPerBlk - 1) / rowsPerBlk;
    const int gGemm = (N + 127) / 128;
    const float invN = 1.0f / (float)N;

    // conv1 (no BN)
    embed_kernel<<<gNC, bThr, 0, stream>>>(inputs, W_in, b_in, X, N);

    for (int i = 0; i < NLAYERS; ++i) {
        const bool lap = (i % 2 == 0);
        const float* W0i = W0 + (size_t)i * 2 * CH * CH;
        const float* b0i = b0 + (size_t)i * CH;
        const float* g0i = g0 + (size_t)i * 2 * CH;
        const float* t0i = bt0 + (size_t)i * 2 * CH;
        const float* W1i = W1 + (size_t)i * 2 * CH * CH;
        const float* b1i = b1 + (size_t)i * CH;
        const float* g1i = g1 + (size_t)i * 2 * CH;
        const float* t1i = bt1 + (size_t)i * 2 * CH;

        for (int half = 0; half < 2; ++half) {
            const float* src = (half == 0) ? X : Yb;   // pre-activation input; elu fused below

            if (lap) {
                zero_f32<<<gNC, bThr, 0, stream>>>(oph, NC);
                long long tot = (long long)NEDGE * CH;
                int gE = (int)((tot + bThr - 1) / bThr);
                spmm_edges<<<gE, bThr, 0, stream>>>(Lrows, Lcols, Lvals, src, oph, tot);
            } else {
                zero_f32<<<1, 256, 0, stream>>>(avgbuf, CH + 1);
                avg_reduce<<<gRows, CH, 0, stream>>>(src, mask, avgbuf, N, rowsPerBlk);
                avg_bcast<<<gNC, bThr, 0, stream>>>(avgbuf, oph, NC);
            }

            zero_f32<<<2, 256, 0, stream>>>(stats, 2 * 2 * CH);
            pack_stats<<<gRows, 256, 0, stream>>>(src, oph, A16, stats, N, 2 * CH, rowsPerBlk);

            if (half == 0) {
                fold_weights<<<1, CH, 0, stream>>>(W0i, b0i, g0i, t0i, stats, Wt16, biasBuf,
                                                   2 * CH, CH, invN);
                gemm_wmma<<<gGemm, 256, 0, stream>>>(A16, Wt16, biasBuf, nullptr, nullptr,
                                                     Yb, N, 2 * CH, CH, CH);
            } else {
                fold_weights<<<1, CH, 0, stream>>>(W1i, b1i, g1i, t1i, stats, Wt16, biasBuf,
                                                   2 * CH, CH, invN);
                // residual: X = gemm + X
                gemm_wmma<<<gGemm, 256, 0, stream>>>(A16, Wt16, biasBuf, X, nullptr,
                                                     X, N, 2 * CH, CH, CH);
            }
        }
    }

    // final: elu -> BN(pre) -> @W_out + b_out + tile(inputs, 40)
    zero_f32<<<1, 256, 0, stream>>>(stats, 2 * CH);
    pack_stats<<<gRows, 256, 0, stream>>>(X, nullptr, A16, stats, N, CH, rowsPerBlk);
    fold_weights<<<1, CH, 0, stream>>>(W_out, b_out, g_out, bt_out, stats, Wt16, biasBuf,
                                       CH, 120, invN);
    gemm_wmma<<<gGemm, 256, 0, stream>>>(A16, Wt16, biasBuf, nullptr, inputs,
                                         out, N, CH, 120, 120);
}
